// GraphAttentionLayer_40931038331168
// MI455X (gfx1250) — matrix operand order repaired and checked by reading, not compiled, not run on hardware
//
#include <hip/hip_runtime.h>
#include <hip/hip_bf16.h>

// ---------- problem constants ----------
#define BB 4
#define NN 2048
#define FIN 256
#define HH 4
#define DD 64

// ---------- vector types ----------
typedef __bf16 bf16_t;
typedef bf16_t v16bf __attribute__((ext_vector_type(16)));
typedef bf16_t v8bf  __attribute__((ext_vector_type(8)));
typedef float  v8f   __attribute__((ext_vector_type(8)));
typedef unsigned int   uint_t;
typedef unsigned short ushort_t;
typedef uint_t   u32x4  __attribute__((ext_vector_type(4)));
typedef ushort_t u16x16 __attribute__((ext_vector_type(16)));

// ---------- helpers ----------
__device__ __forceinline__ ushort_t f2bf(float f) {
    uint_t u = __float_as_uint(f);
    u += 0x7fffu + ((u >> 16) & 1u);          // round-to-nearest-even
    return (ushort_t)(u >> 16);
}
__device__ __forceinline__ float bf2f(ushort_t s) {
    return __uint_as_float(((uint_t)s) << 16);
}
__device__ __forceinline__ float bperm_f(int src_lane, float v) {
    return __int_as_float(__builtin_amdgcn_ds_bpermute(src_lane << 2, __float_as_int(v)));
}

// ---------- prep: W[h][k][d] (f32) -> WT[h][d][k] (bf16) ----------
__global__ void prep_w(const float* __restrict__ W, ushort_t* __restrict__ WT) {
    int i = blockIdx.x * 256 + threadIdx.x;        // over HH*DD*FIN = 65536
    if (i >= HH * DD * FIN) return;
    int k = i & (FIN - 1);
    int d = (i >> 8) & (DD - 1);
    int h = i >> 14;
    WT[i] = f2bf(W[(h * FIN + k) * DD + d]);
}

// ---------- prep: x (f32) -> xb (bf16), same layout ----------
__global__ void prep_x(const float* __restrict__ x, ushort_t* __restrict__ xb) {
    int i = blockIdx.x * 256 + threadIdx.x;        // over BB*NN*FIN = 2M
    if (i >= BB * NN * FIN) return;
    xb[i] = f2bf(x[i]);
}

// ---------- projection GEMM: hT[b,h,d,n] = sum_k x[b,n,k] * W[h,k,d]  (bf16 WMMA) ----------
__global__ __launch_bounds__(128) void gat_proj(const bf16_t* __restrict__ xb,
                                                const bf16_t* __restrict__ WT,
                                                ushort_t* __restrict__ hT) {
    const int tid  = threadIdx.x;
    const int ln   = tid & 31;
    const int half = ln >> 4;
    const int lr   = ln & 15;
    const int wid  = blockIdx.x * 4 + (tid >> 5);   // 2048 waves total
    const int b     = wid >> 9;          // HH*128 = 512 waves per batch
    const int head  = (wid >> 7) & 3;
    const int mtile = wid & 127;

    v8f acc[4];
#pragma unroll
    for (int i = 0; i < 4; ++i)
#pragma unroll
        for (int r = 0; r < 8; ++r) acc[i][r] = 0.0f;

    const bf16_t* xrow = xb + ((size_t)(b * NN + mtile * 16 + lr)) * FIN;
    const bf16_t* wb   = WT + (size_t)head * DD * FIN;

#pragma unroll 2
    for (int kc = 0; kc < FIN; kc += 32) {
        // A operand: 16x32 bf16, lane = row, 16 consecutive K
        v8bf lo = *(const v8bf*)(xrow + kc + half * 16);
        v8bf hi = *(const v8bf*)(xrow + kc + half * 16 + 8);
        v16bf aT = __builtin_shufflevector(lo, hi, 0, 1, 2, 3, 4, 5, 6, 7,
                                           8, 9, 10, 11, 12, 13, 14, 15);
#pragma unroll
        for (int d4 = 0; d4 < 4; ++d4) {
            // B operand: 32x16 bf16, lane = column d, 16 consecutive K (WT is k-contiguous)
            v16bf bT = *(const v16bf*)(wb + (size_t)(d4 * 16 + lr) * FIN + kc + half * 16);
            acc[d4] = __builtin_amdgcn_wmma_f32_16x16x32_bf16(
                false, aT, false, bT, (short)0, acc[d4], false, false);
        }
    }

    const int bh = b * HH + head;
#pragma unroll
    for (int d4 = 0; d4 < 4; ++d4)
#pragma unroll
        for (int r = 0; r < 8; ++r) {
            int n = mtile * 16 + r + half * 8;     // C layout: M = r + 8*half
            int d = d4 * 16 + lr;                  // C layout: N = lane%16
            hT[((size_t)(bh * DD + d)) * NN + n] = f2bf(acc[d4][r]);
        }
}

// ---------- attention logit pieces: e_i = h.a1 , e_j = h.a2 ----------
__global__ void calc_e(const ushort_t* __restrict__ hT, const float* __restrict__ a,
                       float* __restrict__ ei, float* __restrict__ ej) {
    int i = blockIdx.x * 256 + threadIdx.x;        // over BB*HH*NN = 32768
    if (i >= BB * HH * NN) return;
    int n  = i & (NN - 1);
    int bh = i >> 11;
    int h  = bh & (HH - 1);
    float s1 = 0.f, s2 = 0.f;
#pragma unroll 4
    for (int d = 0; d < DD; ++d) {
        float hv = bf2f(hT[((size_t)(bh * DD + d)) * NN + n]);
        s1 += hv * a[h * 2 * DD + d];
        s2 += hv * a[h * 2 * DD + DD + d];
    }
    ei[i] = s1;
    ej[i] = s2;
}

// ---------- fused masked-softmax + PV (flash style), 4 heads share one adj tile ----------
__global__ __launch_bounds__(128) void gat_attn(const int* __restrict__ adj,
                                                const bf16_t* __restrict__ hT,
                                                const float* __restrict__ ei,
                                                const float* __restrict__ ej,
                                                float* __restrict__ out) {
    __shared__ int s_adj[16 * 32];

    const int tid  = threadIdx.x;
    const int head = tid >> 5;          // wave == head
    const int ln   = tid & 31;
    const int half = ln >> 4;
    const int lr   = ln & 15;
    const int b     = blockIdx.x >> 7;  // grid = BB * (NN/16)
    const int mtile = blockIdx.x & 127;
    const int bh    = b * HH + head;

    const int*   adj_rowbase = adj + ((size_t)(b * NN + mtile * 16)) * NN;
    const float  my_ei = ei[bh * NN + mtile * 16 + lr];
    const bf16_t* vbase = hT + (size_t)bh * DD * NN;

    v8f acc[4];
#pragma unroll
    for (int i = 0; i < 4; ++i)
#pragma unroll
        for (int r = 0; r < 8; ++r) acc[i][r] = 0.0f;

    float m_run = -1e30f;
    float l_run = 0.0f;

    for (int chunk = 0; chunk < NN; chunk += 32) {
        // ---- cooperative adj tile load (16x32 ints, shared by all 4 head-waves) ----
        __syncthreads();
        {
            int f = tid * 4;                      // 128 threads * 4 ints = 512
            int r = f >> 5, c = f & 31;
            const int* g = adj_rowbase + (size_t)r * NN + chunk + c;
            *(u32x4*)&s_adj[f] = *(const u32x4*)g;
            if (chunk + 32 < NN) __builtin_prefetch(g + 32, 0, 1);   // global_prefetch next tile
        }
        __syncthreads();

        float ejreg = ej[bh * NN + chunk + ln];   // lane k holds e_j[chunk+k]

        // ---- scores in WMMA-A register layout (lane = row lr, 16 K values) ----
        float s[16];
        float mloc = -1e30f;
#pragma unroll
        for (int v = 0; v < 8; ++v) {
            int K0 = half * 16 + 2 * v;
            float e0 = bperm_f(K0, ejreg);
            float e1 = bperm_f(K0 + 1, ejreg);
            int a0 = s_adj[lr * 32 + K0];
            int a1 = s_adj[lr * 32 + K0 + 1];
            float s0 = my_ei + e0; s0 = (s0 > 0.f) ? s0 : 0.2f * s0;
            float s1 = my_ei + e1; s1 = (s1 > 0.f) ? s1 : 0.2f * s1;
            s0 = (a0 == 0) ? -1e30f : s0;
            s1 = (a1 == 0) ? -1e30f : s1;
            s[2 * v] = s0; s[2 * v + 1] = s1;
            mloc = fmaxf(mloc, fmaxf(s0, s1));
        }
        // ---- online softmax (rows split across lane pairs ln, ln^16) ----
        float mrow  = fmaxf(mloc, bperm_f(ln ^ 16, mloc));
        float m_new = fmaxf(m_run, mrow);
        float fac   = __expf(m_run - m_new);
        m_run = m_new;

        float psum = 0.f;
        u16x16 pp;
#pragma unroll
        for (int e = 0; e < 16; ++e) {
            float p = __expf(s[e] - m_new);
            psum += p;
            pp[e] = f2bf(p);
        }
        psum += bperm_f(ln ^ 16, psum);
        l_run = l_run * fac + psum;

        // factor per C-layout row (VGPR r -> row r + 8*half)
        v8f facv;
#pragma unroll
        for (int r = 0; r < 8; ++r) facv[r] = bperm_f(r + (half << 3), fac);

        v16bf aP = __builtin_bit_cast(v16bf, pp);
#pragma unroll
        for (int d4 = 0; d4 < 4; ++d4) {
            acc[d4] *= facv;
            // B operand: V chunk 32x16, lane = column d, 16 consecutive source nodes (hT n-contiguous)
            v16bf vB = *(const v16bf*)(vbase + (size_t)(d4 * 16 + lr) * NN + chunk + half * 16);
            acc[d4] = __builtin_amdgcn_wmma_f32_16x16x32_bf16(
                false, aP, false, vB, (short)0, acc[d4], false, false);
        }
    }

    // ---- normalize and store ----
    float inv = 1.0f / l_run;
    v8f invv;
#pragma unroll
    for (int r = 0; r < 8; ++r) invv[r] = bperm_f(r + (half << 3), inv);
#pragma unroll
    for (int d4 = 0; d4 < 4; ++d4) {
        acc[d4] *= invv;
#pragma unroll
        for (int r = 0; r < 8; ++r) {
            int n = mtile * 16 + r + half * 8;
            out[((size_t)(b * NN + n)) * (HH * DD) + head * DD + d4 * 16 + lr] = acc[d4][r];
        }
    }
}

// ---------- launcher ----------
extern "C" void kernel_launch(void* const* d_in, const int* in_sizes, int n_in,
                              void* d_out, int out_size, void* d_ws, size_t ws_size,
                              hipStream_t stream) {
    const float* x   = (const float*)d_in[0];
    const int*   adj = (const int*)d_in[1];
    const float* W   = (const float*)d_in[2];
    const float* a   = (const float*)d_in[3];
    float* out = (float*)d_out;

    char* ws = (char*)d_ws;
    size_t off = 0;
    ushort_t* WT = (ushort_t*)(ws + off); off += (size_t)HH * DD * FIN * 2;   // 128 KB
    ushort_t* xb = (ushort_t*)(ws + off); off += (size_t)BB * NN * FIN * 2;   // 4 MB
    ushort_t* hT = (ushort_t*)(ws + off); off += (size_t)BB * HH * DD * NN * 2; // 4 MB
    float*    ei = (float*)(ws + off);    off += (size_t)BB * HH * NN * 4;    // 128 KB
    float*    ej = (float*)(ws + off);    off += (size_t)BB * HH * NN * 4;    // 128 KB
    (void)ws_size; (void)in_sizes; (void)n_in; (void)out_size;

    prep_w<<<(HH * DD * FIN + 255) / 256, 256, 0, stream>>>(W, WT);
    prep_x<<<(BB * NN * FIN + 255) / 256, 256, 0, stream>>>(x, xb);
    gat_proj<<<(BB * HH * (NN / 16)) / 4, 128, 0, stream>>>(
        (const bf16_t*)xb, (const bf16_t*)WT, hT);
    calc_e<<<(BB * HH * NN + 255) / 256, 256, 0, stream>>>(hT, a, ei, ej);
    gat_attn<<<BB * (NN / 16), 128, 0, stream>>>(
        adj, (const bf16_t*)hT, ei, ej, out);
}
